// HandwritingSynthesisNetwork_37735582662868
// MI455X (gfx1250) — matrix operand order repaired and checked by reading, not compiled, not run on hardware
//
#include <hip/hip_runtime.h>
#include <math.h>

// ---------------------------------------------------------------------------
// Handwriting synthesis net (Graves) for MI455X / gfx1250.
// B=64, T=600, HID=400, K=10, U=64, D=60.
// bf16 WMMA (v_wmma_f32_16x16x32_bf16) for every GEMM, fp32 cell state.
// WMMA k-loops are software-pipelined per segment with pure pointer-increment
// addressing: steady state = 3 adds + 6 b128 loads + prefetch + 2 wmma.
// ---------------------------------------------------------------------------

typedef __attribute__((ext_vector_type(16))) __bf16 bf16x16;
typedef __attribute__((ext_vector_type(8)))  float  v8f;

#define HIDW   416        // 400 padded to 13*32
#define GROUP  (64*HIDW)  // one timestep group of H history (64 rows x 416)

__device__ __forceinline__ unsigned short f2bf(float f) {
  unsigned int u = __float_as_uint(f);
  unsigned int r = (u + 0x7FFFu + ((u >> 16) & 1u)) >> 16;   // RNE
  return (unsigned short)r;
}

union FragU { bf16x16 v; uint4 q[2]; };

// A fragment (16x32 bf16): lane l = row m0+(l&15); halves 0..7 at
// K=k0+(l>=16?8:0)+0..7, halves 8..15 at +16..23  -> two b128 loads, +32B apart.
__device__ __forceinline__ bf16x16 load_fragA(const unsigned short* p) {
  const uint4* q = (const uint4*)p;
  FragU r; r.q[0] = q[0]; r.q[1] = q[2];
  return r.v;
}
// B fragment (32x16 bf16): lane l = W row n0+(l&15); halves 0..7 at
// K=k0+(l>=16?8:0)+0..7, halves 8..15 at +16..23  -> two b128 loads, +32B apart.
__device__ __forceinline__ bf16x16 load_fragB(const unsigned short* p) {
  const uint4* q = (const uint4*)p;
  FragU r; r.q[0] = q[0]; r.q[1] = q[2];
  return r.v;
}
__device__ __forceinline__ const unsigned short* aptr(const unsigned short* base, int stride,
                                                      int m0, int lane) {
  return base + (m0 + (lane & 15)) * stride + ((lane >> 4) << 3);
}
__device__ __forceinline__ const unsigned short* bptr(const unsigned short* base, int stride,
                                                      int n0, int lane) {
  return base + (n0 + (lane & 15)) * stride + ((lane >> 4) << 3);
}

__device__ __forceinline__ v8f wmma_bf16(bf16x16 a, bf16x16 b, v8f c) {
  return __builtin_amdgcn_wmma_f32_16x16x32_bf16(false, a, false, b, (short)0, c, false, false);
}

// Pipelined GEMM over one K segment, two B streams (two gates), pointer-
// increment addressing only.
__device__ __forceinline__ void gemm_seg2(const unsigned short* A, int sa,
                                          const unsigned short* B, int sb, int ks,
                                          int m0, int wr0, int wr1, int lane,
                                          v8f& acc0, v8f& acc1) {
  const unsigned short* pa  = aptr(A, sa, m0, lane);
  const unsigned short* pb0 = bptr(B, sb, wr0, lane);
  const unsigned short* pb1 = bptr(B, sb, wr1, lane);
  bf16x16 a  = load_fragA(pa);
  bf16x16 b0 = load_fragB(pb0);
  bf16x16 b1 = load_fragB(pb1);
  for (int k = 0; k < ks - 1; ++k) {
    pa += 32; pb0 += 32; pb1 += 32;
    __builtin_prefetch((const void*)(pb0 + 32), 0, 1);   // global_prefetch_b8
    bf16x16 an  = load_fragA(pa);
    bf16x16 b0n = load_fragB(pb0);
    bf16x16 b1n = load_fragB(pb1);
    acc0 = wmma_bf16(a, b0, acc0);    // overlaps the loads above
    acc1 = wmma_bf16(a, b1, acc1);
    a = an; b0 = b0n; b1 = b1n;
  }
  acc0 = wmma_bf16(a, b0, acc0);
  acc1 = wmma_bf16(a, b1, acc1);
}

// Single-B-stream variant.
__device__ __forceinline__ void gemm_seg1(const unsigned short* A, int sa,
                                          const unsigned short* B, int sb, int ks,
                                          int m0, int n0, int lane, v8f& acc) {
  const unsigned short* pa = aptr(A, sa, m0, lane);
  const unsigned short* pb = bptr(B, sb, n0, lane);
  bf16x16 a = load_fragA(pa);
  bf16x16 b = load_fragB(pb);
  for (int k = 0; k < ks - 1; ++k) {
    pa += 32; pb += 32;
    bf16x16 an = load_fragA(pa);
    bf16x16 bn = load_fragB(pb);
    acc = wmma_bf16(a, b, acc);
    a = an; b = bn;
  }
  acc = wmma_bf16(a, b, acc);
}

__device__ __forceinline__ float sigm(float x) { return 1.f / (1.f + expf(-x)); }

// ---------------------------------------------------------------------------
// Prep: convert weights fp32 -> padded bf16, init states/history buffers.
// ---------------------------------------------------------------------------
__global__ void hw_prep(const float* __restrict__ strokes, const float* __restrict__ onehots,
                        const float* __restrict__ Wih1f, const float* __restrict__ Whh1f,
                        const float* __restrict__ Wwinf, const float* __restrict__ Wih2f,
                        const float* __restrict__ Whh2f, const float* __restrict__ Wfcf,
                        unsigned short* Wih1, unsigned short* Whh1,
                        unsigned short* Wih2a, unsigned short* Wih2b, unsigned short* Whh2,
                        unsigned short* Wfca, unsigned short* Wfcb, unsigned short* Wwin,
                        unsigned short* Xw1, unsigned short* XW2,
                        unsigned short* H1a, unsigned short* H2a,
                        float* Cb1, float* Cb2, float* hc1, float* hc2,
                        float* kappa, float* wcur) {
  int tid = blockIdx.x * blockDim.x + threadIdx.x;
  int st  = gridDim.x * blockDim.x;
  for (int i = tid; i < 1600 * 64; i += st) { int n = i >> 6, c = i & 63;
    Wih1[i]  = f2bf(c < 63 ? Wih1f[n * 63 + c] : 0.f); }
  for (int i = tid; i < 1600 * HIDW; i += st) { int n = i / HIDW, c = i % HIDW;
    Whh1[i]  = f2bf(c < 400 ? Whh1f[n * 400 + c] : 0.f); }
  for (int i = tid; i < 1600 * 64; i += st) { int n = i >> 6, c = i & 63;
    Wih2a[i] = f2bf(c < 63 ? Wih2f[n * 463 + c] : 0.f); }
  for (int i = tid; i < 1600 * HIDW; i += st) { int n = i / HIDW, c = i % HIDW;
    Wih2b[i] = f2bf(c < 400 ? Wih2f[n * 463 + 63 + c] : 0.f); }
  for (int i = tid; i < 1600 * HIDW; i += st) { int n = i / HIDW, c = i % HIDW;
    Whh2[i]  = f2bf(c < 400 ? Whh2f[n * 400 + c] : 0.f); }
  for (int i = tid; i < 128 * HIDW; i += st) { int n = i / HIDW, c = i % HIDW;
    Wfca[i]  = f2bf((n < 121 && c < 400) ? Wfcf[n * 800 + c] : 0.f); }
  for (int i = tid; i < 128 * HIDW; i += st) { int n = i / HIDW, c = i % HIDW;
    Wfcb[i]  = f2bf((n < 121 && c < 400) ? Wfcf[n * 800 + 400 + c] : 0.f); }
  for (int i = tid; i < 32 * HIDW; i += st) { int n = i / HIDW, c = i % HIDW;
    Wwin[i]  = f2bf((n < 30 && c < 400) ? Wwinf[n * 400 + c] : 0.f); }
  for (int i = tid; i < 64 * 64; i += st) { int b = i >> 6, c = i & 63;
    float v = (c < 3) ? strokes[(b * 600) * 3 + c]
            : (c < 63) ? onehots[(b * 64) * 60 + (c - 3)] : 0.f;
    Xw1[i] = f2bf(v); }
  for (int i = tid; i < 38400 * 64; i += st) { int r = i >> 6, c = i & 63;
    int t = r >> 6, b = r & 63;
    XW2[i] = (c < 3) ? f2bf(strokes[(b * 600 + t) * 3 + c]) : (unsigned short)0; }
  for (int i = tid; i < 601 * GROUP; i += st) { H1a[i] = 0; H2a[i] = 0; }
  for (int i = tid; i < 25600; i += st) { Cb1[i] = 0.f; Cb2[i] = 0.f; hc1[i] = 0.f; hc2[i] = 0.f; }
  for (int i = tid; i < 640; i += st) kappa[i] = 0.f;
  for (int i = tid; i < 3840; i += st) { int b = i / 60, d = i % 60;
    wcur[i] = onehots[(b * 64) * 60 + d]; }
}

// ---------------------------------------------------------------------------
// Generic LSTM cell step. 25 blocks; block nb owns hidden cols
// [16*nb,16*nb+16). 8 waves: wave = g0*4 + mtile; wave computes gates g0 and
// g0+2 for its M-tile. Up to 3 K segments (x-part, h1-part, h2-part).
// ---------------------------------------------------------------------------
__global__ __launch_bounds__(256) void hw_lstm_cell(
    const unsigned short* A1, int sa1, int ks1, const unsigned short* B1,
    const unsigned short* A2, int sa2, int ks2, const unsigned short* B2,
    const unsigned short* A3, int sa3, int ks3, const unsigned short* B3,
    const float* __restrict__ bias, float* __restrict__ cbuf,
    float* __restrict__ hfp, unsigned short* __restrict__ hdst) {
  __shared__ float gbuf[4 * 64 * 16];
  int lane = threadIdx.x & 31, wave = threadIdx.x >> 5;
  int mt = wave & 3, g0 = wave >> 2;
  int nb = blockIdx.x * 16;
  int m0 = mt * 16;
  int wr0 = g0 * 400 + nb, wr1 = (g0 + 2) * 400 + nb;
  v8f acc0 = {}, acc1 = {};

  gemm_seg2(A1, sa1, B1, ks1 * 32, ks1, m0, wr0, wr1, lane, acc0, acc1);
  gemm_seg2(A2, sa2, B2, ks2 * 32, ks2, m0, wr0, wr1, lane, acc0, acc1);
  if (ks3) gemm_seg2(A3, sa3, B3, ks3 * 32, ks3, m0, wr0, wr1, lane, acc0, acc1);

  float bia0 = bias[wr0 + (lane & 15)];
  float bia1 = bias[wr1 + (lane & 15)];
  int N = lane & 15;
  int Mb = mt * 16 + ((lane >> 4) << 3);
#pragma unroll
  for (int v = 0; v < 8; ++v) {
    gbuf[g0 * 1024 + (Mb + v) * 16 + N]       = acc0[v] + bia0;
    gbuf[(g0 + 2) * 1024 + (Mb + v) * 16 + N] = acc1[v] + bia1;
  }
  __syncthreads();

  for (int e = threadIdx.x; e < 1024; e += 256) {
    int b = e >> 4, jn = e & 15, j = nb + jn;
    float gi = gbuf[e], gf = gbuf[1024 + e], gg = gbuf[2048 + e], go = gbuf[3072 + e];
    float c_old = cbuf[b * 400 + j];
    float cn = sigm(gf) * c_old + sigm(gi) * tanhf(gg);
    float hn = sigm(go) * tanhf(cn);
    cbuf[b * 400 + j] = cn;
    hfp[b * 400 + j]  = hn;
    hdst[b * HIDW + j] = f2bf(hn);
  }
}

// ---------------------------------------------------------------------------
// Attention window for step t (single WG). p = exp(h @ Wwin^T + b) via
// pipelined WMMA, then kappa update, phi, w = phi @ onehots, next inputs.
// ---------------------------------------------------------------------------
__global__ __launch_bounds__(256) void hw_attn_step(
    int t, const unsigned short* __restrict__ Hb,  // h_t bf16, stride HIDW
    const unsigned short* __restrict__ Wwin, const float* __restrict__ bwin,
    const float* __restrict__ tmask, const float* __restrict__ onehots,
    const float* __restrict__ strokes,
    float* __restrict__ kappa, float* __restrict__ phibuf, float* __restrict__ wcur,
    unsigned short* __restrict__ XW2, unsigned short* __restrict__ Xw1) {
  __shared__ float p_s[64 * 32];
  __shared__ float kap_s[640];
  __shared__ float phi_s[4096];
  int lane = threadIdx.x & 31, wave = threadIdx.x >> 5;

  {
    int mt = wave & 3, nt = wave >> 2;  // nt in 0..1
    v8f acc = {};
    gemm_seg1(Hb, HIDW, Wwin, HIDW, 13, mt * 16, nt * 16, lane, acc);
    int col = nt * 16 + (lane & 15);
    float bia = (col < 30) ? bwin[col] : 0.f;
    int Mb = mt * 16 + ((lane >> 4) << 3);
#pragma unroll
    for (int v = 0; v < 8; ++v) p_s[(Mb + v) * 32 + col] = expf(acc[v] + bia);
  }
  __syncthreads();

  for (int i = threadIdx.x; i < 640; i += 256) {
    int b = i / 10, k = i % 10;
    float nk = kappa[i] + p_s[b * 32 + 20 + k];
    kappa[i] = nk; kap_s[i] = nk;
  }
  __syncthreads();

  for (int i = threadIdx.x; i < 4096; i += 256) {
    int b = i >> 6, u = i & 63;
    float s = 0.f;
#pragma unroll
    for (int k = 0; k < 10; ++k) {
      float d = kap_s[b * 10 + k] - (float)u;
      s += p_s[b * 32 + k] * expf(-p_s[b * 32 + 10 + k] * d * d);
    }
    s *= tmask[i];
    phi_s[i] = s; phibuf[i] = s;
  }
  __syncthreads();

  for (int i = threadIdx.x; i < 3840; i += 256) {
    int b = i / 60, d = i % 60;
    float s = 0.f;
    const float* oh = onehots + (b * 64) * 60 + d;
    for (int u = 0; u < 64; ++u) s += phi_s[b * 64 + u] * oh[u * 60];
    wcur[i] = s;
    unsigned short bv = f2bf(s);
    XW2[(t * 64 + b) * 64 + 3 + d] = bv;
    if (t + 1 < 600) Xw1[b * 64 + 3 + d] = bv;
  }
  if (t + 1 < 600) {
    for (int i = threadIdx.x; i < 192; i += 256) {
      int b = i / 3, c = i % 3;
      Xw1[b * 64 + c] = f2bf(strokes[(b * 600 + t + 1) * 3 + c]);
    }
  }
}

// ---------------------------------------------------------------------------
// Final FC: out = [H1|H2](38400 x 832 bf16) @ Wfc^T + b_fc -> obuf fp32.
// 2400 blocks x 8 waves; wave = N-tile. Two pipelined 13-step segments.
// ---------------------------------------------------------------------------
__global__ __launch_bounds__(256) void hw_fc_gemm(
    const unsigned short* __restrict__ H1, const unsigned short* __restrict__ H2,
    const unsigned short* __restrict__ Wa, const unsigned short* __restrict__ Wb,
    const float* __restrict__ bfc, float* __restrict__ obuf) {
  int lane = threadIdx.x & 31, nt = threadIdx.x >> 5;
  int m0 = blockIdx.x * 16, n0 = nt * 16;
  v8f acc = {};
  gemm_seg1(H1, HIDW, Wa, HIDW, 13, m0, n0, lane, acc);
  gemm_seg1(H2, HIDW, Wb, HIDW, 13, m0, n0, lane, acc);

  int col = n0 + (lane & 15);
  float bia = (col < 121) ? bfc[col] : 0.f;
  int Mb = m0 + ((lane >> 4) << 3);
#pragma unroll
  for (int v = 0; v < 8; ++v) obuf[(Mb + v) * 128 + col] = acc[v] + bia;
}

// ---------------------------------------------------------------------------
// Per-row output transforms + scatter into d_out. Row r = t*64 + b.
// ---------------------------------------------------------------------------
__global__ __launch_bounds__(256) void hw_out_transform(const float* __restrict__ obuf,
                                                        float* __restrict__ dout) {
  int r = blockIdx.x * 256 + threadIdx.x;
  if (r >= 38400) return;
  const float* o = obuf + r * 128;
  int b = r & 63, t = r >> 6;
  int base = b * 600 + t;
  float* mu = dout + 768000 + base * 40;
  float* sg = dout + 2304000 + base * 40;
  float* pi = dout + base * 20;
  float* rh = dout + 3840000 + base * 20;
  for (int c = 0; c < 40; ++c) { mu[c] = o[c]; sg[c] = o[40 + c]; }
  float mx = o[80];
#pragma unroll
  for (int k = 1; k < 20; ++k) mx = fmaxf(mx, o[80 + k]);
  float e[20], s = 0.f;
#pragma unroll
  for (int k = 0; k < 20; ++k) { e[k] = expf(o[80 + k] - mx); s += e[k]; }
  float inv = 1.f / s;
#pragma unroll
  for (int k = 0; k < 20; ++k) pi[k] = e[k] * inv;
#pragma unroll
  for (int k = 0; k < 20; ++k) rh[k] = tanhf(o[100 + k]);
  dout[4608000 + base] = 1.f / (1.f + expf(o[120]));  // sigmoid(-eos)
}

__global__ void hw_finals(const float* wcur, const float* kappa, const float* phibuf,
                          const float* hc1, const float* cb1,
                          const float* hc2, const float* cb2, float* dout) {
  int i = blockIdx.x * blockDim.x + threadIdx.x;
  int st = gridDim.x * blockDim.x;
  for (; i < 110976; i += st) {
    if (i < 3840)       dout[4646400 + i] = wcur[i];
    else if (i < 4480)  dout[4650240 + i - 3840] = kappa[i - 3840];
    else if (i < 8576)  dout[4650880 + i - 4480] = phibuf[i - 4480];
    else {
      int j = i - 8576, blk = j / 25600, o = j % 25600;
      const float* src = (blk == 0) ? hc1 : (blk == 1) ? cb1 : (blk == 2) ? hc2 : cb2;
      dout[4654976 + j] = src[o];
    }
  }
}

// ---------------------------------------------------------------------------
extern "C" void kernel_launch(void* const* d_in, const int* in_sizes, int n_in,
                              void* d_out, int out_size, void* d_ws, size_t ws_size,
                              hipStream_t stream) {
  const float* strokes = (const float*)d_in[0];
  const float* onehots = (const float*)d_in[1];
  const float* tmask   = (const float*)d_in[2];
  const float* Wih1f   = (const float*)d_in[3];
  const float* Whh1f   = (const float*)d_in[4];
  const float* b1      = (const float*)d_in[5];
  const float* Wwinf   = (const float*)d_in[6];
  const float* bwin    = (const float*)d_in[7];
  const float* Wih2f   = (const float*)d_in[8];
  const float* Whh2f   = (const float*)d_in[9];
  const float* b2      = (const float*)d_in[10];
  const float* Wfcf    = (const float*)d_in[11];
  const float* bfc     = (const float*)d_in[12];
  float* dout = (float*)d_out;

  // Workspace layout (bf16 region, all 16B aligned, then fp32 region). ~90MB.
  unsigned short* p = (unsigned short*)d_ws;
  unsigned short* Wih1  = p; p += 1600 * 64;
  unsigned short* Whh1  = p; p += 1600 * HIDW;
  unsigned short* Wih2a = p; p += 1600 * 64;
  unsigned short* Wih2b = p; p += 1600 * HIDW;
  unsigned short* Whh2  = p; p += 1600 * HIDW;
  unsigned short* Wfca  = p; p += 128 * HIDW;
  unsigned short* Wfcb  = p; p += 128 * HIDW;
  unsigned short* Wwin  = p; p += 32 * HIDW;
  unsigned short* Xw1   = p; p += 64 * 64;
  unsigned short* XW2   = p; p += 38400 * 64;
  unsigned short* H1a   = p; p += 601 * GROUP;   // group g = h_{g-1}; group 0 = zeros
  unsigned short* H2a   = p; p += 601 * GROUP;
  float* f = (float*)p;
  float* Cb1   = f; f += 25600;
  float* Cb2   = f; f += 25600;
  float* hc1   = f; f += 25600;
  float* hc2   = f; f += 25600;
  float* kappa = f; f += 640;
  float* wcur  = f; f += 3840;
  float* phib  = f; f += 4096;
  float* obuf  = f; f += 38400 * 128;
  (void)in_sizes; (void)n_in; (void)out_size; (void)ws_size;

  hw_prep<<<2048, 256, 0, stream>>>(strokes, onehots, Wih1f, Whh1f, Wwinf, Wih2f, Whh2f, Wfcf,
                                    Wih1, Whh1, Wih2a, Wih2b, Whh2, Wfca, Wfcb, Wwin,
                                    Xw1, XW2, H1a, H2a, Cb1, Cb2, hc1, hc2, kappa, wcur);

  // ---- scan 1: LSTM1 + attention window ----
  for (int t = 0; t < 600; ++t) {
    hw_lstm_cell<<<25, 256, 0, stream>>>(
        Xw1, 64, 2, Wih1,
        H1a + (size_t)t * GROUP, HIDW, 13, Whh1,
        (const unsigned short*)nullptr, 0, 0, (const unsigned short*)nullptr,
        b1, Cb1, hc1, H1a + (size_t)(t + 1) * GROUP);
    hw_attn_step<<<1, 256, 0, stream>>>(
        t, H1a + (size_t)(t + 1) * GROUP, Wwin, bwin, tmask, onehots, strokes,
        kappa, phib, wcur, XW2, Xw1);
  }

  // ---- scan 2: LSTM2 (x2 = [stroke,w | h1] + recurrent h2) ----
  for (int t = 0; t < 600; ++t) {
    hw_lstm_cell<<<25, 256, 0, stream>>>(
        XW2 + (size_t)t * 64 * 64, 64, 2, Wih2a,
        H1a + (size_t)(t + 1) * GROUP, HIDW, 13, Wih2b,
        H2a + (size_t)t * GROUP, HIDW, 13, Whh2,
        b2, Cb2, hc2, H2a + (size_t)(t + 1) * GROUP);
  }

  // ---- parallel tail ----
  hw_fc_gemm<<<2400, 256, 0, stream>>>(H1a + GROUP, H2a + GROUP, Wfca, Wfcb, bfc, obuf);
  hw_out_transform<<<150, 256, 0, stream>>>(obuf, dout);
  hw_finals<<<128, 256, 0, stream>>>(wcur, kappa, phib, hc1, Cb1, hc2, Cb2, dout);
}
